// StormerViT_61795989454887
// MI455X (gfx1250) — compile-verified, matrix operand order checked by reading, not run on hardware
//
#include <hip/hip_runtime.h>
#include <hip/hip_bf16.h>
#include <math.h>

// ---------------- types ----------------
typedef _Float16 half_t;
typedef __attribute__((ext_vector_type(2)))  _Float16 h2_t;
typedef __attribute__((ext_vector_type(16))) _Float16 v16h;
typedef __attribute__((ext_vector_type(8)))  float    v8f;
typedef __attribute__((ext_vector_type(4)))  unsigned int u32x4;
typedef __attribute__((ext_vector_type(8)))  int      i32x8;
typedef __attribute__((ext_vector_type(4)))  int      i32x4;

// ---------------- model dims ----------------
#define N_TOK   8192
#define DMODEL  1024
#define HEADS   8
#define HD      128
#define MLP_H   4096
#define TRI     3072      // 3*D
#define KP      280       // C_IN*P*P
#define KP_PAD  288       // padded to multiple of 32
#define C_OUT   69
#define HN      276       // C_OUT*P*P
#define HN_PAD  384       // padded to multiple of 128
#define IMG_H   128
#define IMG_W   256
#define NW      128       // IMG_W/2

// -------- Tensor Data Mover availability --------
#if defined(__HIP_DEVICE_COMPILE__) && __has_builtin(__builtin_amdgcn_tensor_load_to_lds) && __has_builtin(__builtin_amdgcn_s_wait_tensorcnt)
#define HAVE_TDM 1
#else
#define HAVE_TDM 0
#endif

#if HAVE_TDM
#if defined(__clang_major__) && __clang_major__ >= 23
#define TDM_CALL(g0, g1, g2, g3) __builtin_amdgcn_tensor_load_to_lds((g0), (g1), (g2), (g3), (i32x8){}, 0)
#else
#define TDM_CALL(g0, g1, g2, g3) __builtin_amdgcn_tensor_load_to_lds((g0), (g1), (g2), (g3), 0)
#endif

__device__ __forceinline__ unsigned lds_addr32(const void* p) {
    // ISA: generic LDS address low 32 bits == LDS byte offset
    return (unsigned)(uintptr_t)p;
}

// Issue one 2D TDM tile load: tile_h rows x tile_w f16 elements, global row
// stride stride_el elements, optional LDS row padding (ivl/pad codes per D#).
// ivl code: 3 -> pad every 16 DWORDs, 5 -> every 64 DWORDs; pad code 3 -> +4 DWORDs.
__device__ __forceinline__ void tdm_load_2d(unsigned lds_off, const half_t* g,
                                            int tile_w, int tile_h, long stride_el,
                                            int ivl_code, int pad_code) {
    unsigned long long ga = (unsigned long long)(uintptr_t)g;
    u32x4 g0;
    g0[0] = 1u;                                   // count=1 valid descriptor
    g0[1] = lds_off;                              // lds_addr
    g0[2] = (unsigned)ga;                         // global_addr[31:0]
    g0[3] = (unsigned)((ga >> 32) & 0x01ffffffu)  // global_addr[56:32]
            | 0x80000000u;                        // type=2 ("image")
    i32x8 g1;
    g1[0] = (int)((1u << 16)                      // data_size = 2 bytes
            | (1u << 20)                          // pad_enable
            | ((unsigned)ivl_code << 22)          // pad_interval
            | ((unsigned)pad_code << 25));        // pad_amount
    g1[1] = (int)((unsigned)tile_w << 16);        // tensor_dim0 = tile_w (no OOB)
    g1[2] = (int)((unsigned)tile_h << 16);        // tensor_dim1 = tile_h
    g1[3] = (int)((unsigned)tile_w << 16);        // tile_dim0
    g1[4] = tile_h;                               // tile_dim1 (tile_dim2=0)
    g1[5] = (int)(unsigned)stride_el;             // tensor_dim0_stride lo32
    g1[6] = (int)(((unsigned long long)stride_el >> 32) & 0xffffu);
    g1[7] = 0;
    i32x4 z4 = {0, 0, 0, 0};
    TDM_CALL(g0, g1, z4, z4);
}
#endif // HAVE_TDM

// K index inside a 16-bit WMMA A/B fragment (ISA 7.12.2)
__device__ __forceinline__ int frag_kb(int j, int hi) {
    return (j < 4) ? (hi * 8 + 2 * j) : (16 + hi * 8 + 2 * (j - 4));
}

__device__ __forceinline__ v8f wmma_f16(v16h a, v16h b, v8f c) {
    return __builtin_amdgcn_wmma_f32_16x16x32_f16(false, a, false, b,
                                                  (short)0, c, false, false);
}

// ============================================================
// Weight transpose + f32->f16 convert:  out[KPAD][NPAD] = W[n][k]
// ============================================================
__global__ void convert_wT(const float* __restrict__ W, half_t* __restrict__ out,
                           int NO, int NI, int NOpad, int NIpad) {
    long total = (long)NIpad * NOpad;
    for (long idx = blockIdx.x * (long)blockDim.x + threadIdx.x; idx < total;
         idx += (long)gridDim.x * blockDim.x) {
        int k = (int)(idx / NOpad), n = (int)(idx % NOpad);
        float v = (n < NO && k < NI) ? W[(long)n * NI + k] : 0.f;
        out[idx] = (half_t)v;
    }
}

// ============================================================
// im2col for the stride-2 patch embed
// ============================================================
__global__ void im2col_patch(const float* __restrict__ x, half_t* __restrict__ xp) {
    long total = (long)N_TOK * KP_PAD;
    for (long idx = blockIdx.x * (long)blockDim.x + threadIdx.x; idx < total;
         idx += (long)gridDim.x * blockDim.x) {
        int t = (int)(idx / KP_PAD), col = (int)(idx % KP_PAD);
        float v = 0.f;
        if (col < KP) {
            int c = col >> 2, p = (col >> 1) & 1, q = col & 1;
            int i = t / NW, j = t % NW;
            v = x[(long)c * (IMG_H * IMG_W) + (i * 2 + p) * IMG_W + (j * 2 + q)];
        }
        xp[idx] = (half_t)v;
    }
}

__global__ void add_pos(float* __restrict__ h, const float* __restrict__ pos) {
    long total = (long)N_TOK * DMODEL;
    for (long i = blockIdx.x * (long)blockDim.x + threadIdx.x; i < total;
         i += (long)gridDim.x * blockDim.x)
        h[i] += pos[i];
}

// ============================================================
// LayerNorm: f32 in -> f16 out (one row per block, 128 threads)
// ============================================================
__global__ __launch_bounds__(128)
void ln_kernel(const float* __restrict__ h, const float* __restrict__ w,
               const float* __restrict__ b, half_t* __restrict__ y) {
    __shared__ float r1[128], r2[128];
    int row = blockIdx.x, tid = threadIdx.x;
    float v[8], s = 0.f, s2 = 0.f;
#pragma unroll
    for (int i = 0; i < 8; ++i) {
        v[i] = h[(long)row * DMODEL + tid + i * 128];
        s += v[i]; s2 += v[i] * v[i];
    }
    r1[tid] = s; r2[tid] = s2; __syncthreads();
    for (int st = 64; st > 0; st >>= 1) {
        if (tid < st) { r1[tid] += r1[tid + st]; r2[tid] += r2[tid + st]; }
        __syncthreads();
    }
    float mean = r1[0] * (1.f / DMODEL);
    float var  = r2[0] * (1.f / DMODEL) - mean * mean;
    float rstd = rsqrtf(var + 1e-5f);
#pragma unroll
    for (int i = 0; i < 8; ++i) {
        int col = tid + i * 128;
        y[(long)row * DMODEL + col] = (half_t)((v[i] - mean) * rstd * w[col] + b[col]);
    }
}

// ============================================================
// WMMA GEMM: C[M x N] = A[M x K](f16) * B[K x N](f16) + bias
// block = 256 threads (8 waves), tile 128x128, BK = 32.
// TDM path: double-buffered LDS, wave0 drives the Tensor Data Mover and
// pipelines tile i+1 behind s_wait_tensorcnt(2) while all waves WMMA tile i.
// MODE 0: f32 out = acc+bias      MODE 1: f16 out = acc+bias
// MODE 2: f32 out += acc+bias     MODE 3: f16 out = gelu(acc+bias)
// ============================================================
template <int MODE>
__global__ __launch_bounds__(256)
void gemm_wmma(const half_t* __restrict__ A, const half_t* __restrict__ B,
               const float* __restrict__ bias, int nbias,
               float* __restrict__ outf, half_t* __restrict__ outh,
               int M, int N, int K, int lda, int ldb, int ldo) {
    __shared__ half_t Alds[2][128 * 40];   // row stride 40 (TDM pad 16DW+4DW)
    __shared__ half_t Blds[2][32 * 136];   // row stride 136 (TDM pad 64DW+4DW)

    const int tid = threadIdx.x;
    const int lane = tid & 31, wave = tid >> 5;
    const int hi = lane >> 4, ln = lane & 15;
    const int wm = wave >> 1, wn = wave & 1;      // waves 4x2 over 128x128
    const int m0 = blockIdx.y * 128, n0 = blockIdx.x * 128;
    const int nk = K / 32;

    v8f acc[2][4];
#pragma unroll
    for (int i = 0; i < 2; ++i)
#pragma unroll
        for (int j = 0; j < 4; ++j) acc[i][j] = (v8f){};

#if HAVE_TDM
    if (wave == 0) {   // prologue: tile 0 -> buffer 0
        tdm_load_2d(lds_addr32(&Alds[0][0]), A + (long)m0 * lda, 32, 128, lda, 3, 3);
        tdm_load_2d(lds_addr32(&Blds[0][0]), B + n0, 128, 32, ldb, 5, 3);
    }
#endif

    for (int it = 0; it < nk; ++it) {
#if HAVE_TDM
        const int p = it & 1;
        if (wave == 0) {
            if (it + 1 < nk) {
                tdm_load_2d(lds_addr32(&Alds[p ^ 1][0]),
                            A + (long)m0 * lda + (it + 1) * 32, 32, 128, lda, 3, 3);
                tdm_load_2d(lds_addr32(&Blds[p ^ 1][0]),
                            B + (long)((it + 1) * 32) * ldb + n0, 128, 32, ldb, 5, 3);
                __builtin_amdgcn_s_wait_tensorcnt((short)2);  // tile 'it' retired
            } else {
                __builtin_amdgcn_s_wait_tensorcnt((short)0);
            }
        }
        __syncthreads();
#else
        const int p = 0;
        const int k0 = it * 32;
#pragma unroll
        for (int t = 0; t < 2; ++t) {
            int c = tid + t * 256;
            int ar = c >> 2, ak = (c & 3) * 8;
            *(uint4*)&Alds[0][ar * 40 + ak] =
                *(const uint4*)(A + (long)(m0 + ar) * lda + k0 + ak);
            int bk = c >> 4, bn = (c & 15) * 8;
            *(uint4*)&Blds[0][bk * 136 + bn] =
                *(const uint4*)(B + (long)(k0 + bk) * ldb + n0 + bn);
        }
        if (k0 + 32 < K) {
            __builtin_prefetch(A + (long)(m0 + (tid >> 1)) * lda + k0 + 32, 0, 1);
            __builtin_prefetch(B + (long)(k0 + 32 + (tid >> 3)) * ldb + n0, 0, 1);
        }
        __syncthreads();
#endif
        v16h afr[2];
#pragma unroll
        for (int mt = 0; mt < 2; ++mt) {
            const half_t* ap = &Alds[p][(wm * 32 + mt * 16 + ln) * 40];
#pragma unroll
            for (int j = 0; j < 8; ++j) {
                int kb = frag_kb(j, hi);
                h2_t pr = *(const h2_t*)(ap + kb);
                afr[mt][2 * j] = pr.x; afr[mt][2 * j + 1] = pr.y;
            }
        }
#pragma unroll
        for (int nt = 0; nt < 4; ++nt) {
            v16h bfr;
            int col = wn * 64 + nt * 16 + ln;
#pragma unroll
            for (int j = 0; j < 8; ++j) {
                int kb = frag_kb(j, hi);
                bfr[2 * j]     = Blds[p][kb * 136 + col];
                bfr[2 * j + 1] = Blds[p][(kb + 1) * 136 + col];
            }
#pragma unroll
            for (int mt = 0; mt < 2; ++mt)
                acc[mt][nt] = wmma_f16(afr[mt], bfr, acc[mt][nt]);
        }
        __syncthreads();
    }

    // epilogue: C layout — VGPR r, lanes 0-15: (M=r, N=ln); lanes 16-31: (M=r+8)
#pragma unroll
    for (int mt = 0; mt < 2; ++mt)
#pragma unroll
        for (int nt = 0; nt < 4; ++nt) {
            int gn = n0 + wn * 64 + nt * 16 + ln;
            float bv = (gn < nbias) ? bias[gn] : 0.f;
#pragma unroll
            for (int r = 0; r < 8; ++r) {
                int gm = m0 + wm * 32 + mt * 16 + hi * 8 + r;
                float v = acc[mt][nt][r] + bv;
                long o = (long)gm * ldo + gn;
                if (MODE == 0) outf[o] = v;
                else if (MODE == 1) outh[o] = (half_t)v;
                else if (MODE == 2) outf[o] += v;
                else {
                    float g = 0.5f * v * (1.0f + erff(v * 0.70710678118654752f));
                    outh[o] = (half_t)g;
                }
            }
        }
}

// ============================================================
// Flash attention (f16 in/out, f32 online softmax)
// qkv[t][0..1023]=Q, [1024..2047]=K, [2048..3071]=V; head h -> cols h*128..
// grid = (N/128, HEADS); block = 256 (8 waves, 16 query rows each)
// K/V tiles pulled by the Tensor Data Mover (wave 0), padded to stride 136.
// ============================================================
__global__ __launch_bounds__(256)
void flash_attn(const half_t* __restrict__ qkv, half_t* __restrict__ o16) {
    __shared__ half_t Klds[64 * 136];
    __shared__ half_t Vlds[64 * 136];
    __shared__ half_t Pbuf[8 * 16 * 72];   // per-wave 16x64 P tile, stride 72

    const int tid = threadIdx.x;
    const int lane = tid & 31, wave = tid >> 5;
    const int hi = lane >> 4, ln = lane & 15;
    const int head = blockIdx.y;
    const int qbase = blockIdx.x * 128 + wave * 16;
    const float scale = 0.08838834764831845f;   // HD^-0.5

    // preload this wave's Q rows as 4 A-fragments (K-chunks of 32 over HD=128)
    v16h qa[4];
    {
        const half_t* qp = qkv + (long)(qbase + ln) * TRI + head * HD;
#pragma unroll
        for (int kc = 0; kc < 4; ++kc)
#pragma unroll
            for (int j = 0; j < 8; ++j) {
                int kb = frag_kb(j, hi);
                h2_t p = *(const h2_t*)(qp + kc * 32 + kb);
                qa[kc][2 * j] = p.x; qa[kc][2 * j + 1] = p.y;
            }
    }

    v8f ofr[8];
#pragma unroll
    for (int t = 0; t < 8; ++t) ofr[t] = (v8f){};
    float m8[8], l8[8];
#pragma unroll
    for (int r = 0; r < 8; ++r) { m8[r] = -1e30f; l8[r] = 0.f; }

    half_t* Pw = &Pbuf[wave * 16 * 72];

    for (int kb0 = 0; kb0 < N_TOK; kb0 += 64) {
        __syncthreads();
#if HAVE_TDM
        if (wave == 0) {
            const half_t* base = qkv + (long)kb0 * TRI + head * HD;
            tdm_load_2d(lds_addr32(&Klds[0]), base + DMODEL,     128, 64, TRI, 5, 3);
            tdm_load_2d(lds_addr32(&Vlds[0]), base + 2 * DMODEL, 128, 64, TRI, 5, 3);
            __builtin_amdgcn_s_wait_tensorcnt((short)0);
        }
#else
#pragma unroll
        for (int t = 0; t < 4; ++t) {
            int c = tid + t * 256;
            int row = c >> 4, nc = (c & 15) * 8;
            const half_t* base = qkv + (long)(kb0 + row) * TRI + head * HD + nc;
            *(uint4*)&Klds[row * 136 + nc] = *(const uint4*)(base + DMODEL);
            *(uint4*)&Vlds[row * 136 + nc] = *(const uint4*)(base + 2 * DMODEL);
        }
#endif
        __syncthreads();

        // S = Q * K^T  (4 tiles of 16x16 over 64 keys)
        v8f sfr[4];
#pragma unroll
        for (int nt = 0; nt < 4; ++nt) {
            sfr[nt] = (v8f){};
#pragma unroll
            for (int kc = 0; kc < 4; ++kc) {
                v16h bf;
                const half_t* kp = &Klds[(nt * 16 + ln) * 136 + kc * 32];
#pragma unroll
                for (int j = 0; j < 8; ++j) {
                    int kb = frag_kb(j, hi);
                    h2_t p = *(const h2_t*)(kp + kb);
                    bf[2 * j] = p.x; bf[2 * j + 1] = p.y;
                }
                sfr[nt] = wmma_f16(qa[kc], bf, sfr[nt]);
            }
        }

        // online softmax (rows r+8*hi live in this lane's half-wave)
        float pmax[8];
#pragma unroll
        for (int r = 0; r < 8; ++r) pmax[r] = -1e30f;
#pragma unroll
        for (int nt = 0; nt < 4; ++nt)
#pragma unroll
            for (int r = 0; r < 8; ++r) {
                sfr[nt][r] *= scale;
                pmax[r] = fmaxf(pmax[r], sfr[nt][r]);
            }
#pragma unroll
        for (int off = 1; off < 16; off <<= 1)
#pragma unroll
            for (int r = 0; r < 8; ++r)
                pmax[r] = fmaxf(pmax[r], __shfl_xor(pmax[r], off, 16));

        float alpha[8], psum[8];
#pragma unroll
        for (int r = 0; r < 8; ++r) {
            float mn = fmaxf(m8[r], pmax[r]);
            alpha[r] = __expf(m8[r] - mn);
            m8[r] = mn; psum[r] = 0.f;
        }
#pragma unroll
        for (int nt = 0; nt < 4; ++nt)
#pragma unroll
            for (int r = 0; r < 8; ++r) {
                float p = __expf(sfr[nt][r] - m8[r]);
                sfr[nt][r] = p; psum[r] += p;
            }
#pragma unroll
        for (int off = 1; off < 16; off <<= 1)
#pragma unroll
            for (int r = 0; r < 8; ++r)
                psum[r] += __shfl_xor(psum[r], off, 16);
#pragma unroll
        for (int r = 0; r < 8; ++r) l8[r] = l8[r] * alpha[r] + psum[r];
#pragma unroll
        for (int t = 0; t < 8; ++t)
#pragma unroll
            for (int r = 0; r < 8; ++r) ofr[t][r] *= alpha[r];

        // stage P through LDS: C-layout -> row-major 16x64 (same-wave, in-order)
#pragma unroll
        for (int nt = 0; nt < 4; ++nt)
#pragma unroll
            for (int r = 0; r < 8; ++r)
                Pw[(hi * 8 + r) * 72 + nt * 16 + ln] = (half_t)sfr[nt][r];

        // O += P * V
        v16h pa[2];
#pragma unroll
        for (int kc = 0; kc < 2; ++kc)
#pragma unroll
            for (int j = 0; j < 8; ++j) {
                int kb = frag_kb(j, hi);
                h2_t p = *(const h2_t*)(Pw + ln * 72 + kc * 32 + kb);
                pa[kc][2 * j] = p.x; pa[kc][2 * j + 1] = p.y;
            }
#pragma unroll
        for (int ot = 0; ot < 8; ++ot)
#pragma unroll
            for (int kc = 0; kc < 2; ++kc) {
                v16h vb;
#pragma unroll
                for (int j = 0; j < 8; ++j) {
                    int kb = kc * 32 + frag_kb(j, hi);
                    vb[2 * j]     = Vlds[kb * 136 + ot * 16 + ln];
                    vb[2 * j + 1] = Vlds[(kb + 1) * 136 + ot * 16 + ln];
                }
                ofr[ot] = wmma_f16(pa[kc], vb, ofr[ot]);
            }
    }

    // normalize and store
#pragma unroll
    for (int ot = 0; ot < 8; ++ot)
#pragma unroll
        for (int r = 0; r < 8; ++r) {
            int row = qbase + hi * 8 + r;
            int col = head * HD + ot * 16 + ln;
            o16[(long)row * DMODEL + col] = (half_t)(ofr[ot][r] / l8[r]);
        }
}

// ============================================================
// head output reorder: (N x HN_PAD) f32 -> (C_OUT, 128, 256)
// ============================================================
__global__ void head_reorder(const float* __restrict__ hw, float* __restrict__ out) {
    long total = (long)C_OUT * IMG_H * IMG_W;
    for (long idx = blockIdx.x * (long)blockDim.x + threadIdx.x; idx < total;
         idx += (long)gridDim.x * blockDim.x) {
        int c = (int)(idx / (IMG_H * IMG_W));
        int rem = (int)(idx % (IMG_H * IMG_W));
        int hr = rem / IMG_W, wc = rem % IMG_W;
        int i = hr >> 1, p = hr & 1, j = wc >> 1, q = wc & 1;
        int t = i * NW + j, col = c * 4 + p * 2 + q;
        out[idx] = hw[(long)t * HN_PAD + col];
    }
}

// ============================================================
// host driver
// ============================================================
extern "C" void kernel_launch(void* const* d_in, const int* in_sizes, int n_in,
                              void* d_out, int out_size, void* d_ws, size_t ws_size,
                              hipStream_t stream) {
    const float* x       = (const float*)d_in[0];
    const float* patch_w = (const float*)d_in[1];
    const float* patch_b = (const float*)d_in[2];
    const float* pos     = (const float*)d_in[3];
    const float* ln1_w   = (const float*)d_in[4];
    const float* ln1_b   = (const float*)d_in[5];
    const float* qkv_w   = (const float*)d_in[6];
    const float* qkv_b   = (const float*)d_in[7];
    const float* proj_w  = (const float*)d_in[8];
    const float* proj_b  = (const float*)d_in[9];
    const float* ln2_w   = (const float*)d_in[10];
    const float* ln2_b   = (const float*)d_in[11];
    const float* mlp_w1  = (const float*)d_in[12];
    const float* mlp_b1  = (const float*)d_in[13];
    const float* mlp_w2  = (const float*)d_in[14];
    const float* mlp_b2  = (const float*)d_in[15];
    const float* norm_w  = (const float*)d_in[16];
    const float* norm_b  = (const float*)d_in[17];
    const float* head_w  = (const float*)d_in[18];
    const float* head_b  = (const float*)d_in[19];
    float* out = (float*)d_out;

    char* ws = (char*)d_ws;
    size_t off = 0;
    auto bump = [&](size_t bytes) { void* p = ws + off; off = (off + bytes + 255) & ~(size_t)255; return p; };
    float*  h       = (float*)bump((size_t)N_TOK * DMODEL * 4);
    half_t* y16     = (half_t*)bump((size_t)N_TOK * DMODEL * 2);
    half_t* qkv16   = (half_t*)bump((size_t)N_TOK * TRI * 2);
    half_t* o16     = (half_t*)bump((size_t)N_TOK * DMODEL * 2);
    half_t* g16     = (half_t*)bump((size_t)N_TOK * MLP_H * 2);
    half_t* xp16    = (half_t*)bump((size_t)N_TOK * KP_PAD * 2);
    half_t* wpT     = (half_t*)bump((size_t)KP_PAD * DMODEL * 2);
    half_t* wqkvT   = (half_t*)bump((size_t)2 * DMODEL * TRI * 2);
    half_t* wprojT  = (half_t*)bump((size_t)2 * DMODEL * DMODEL * 2);
    half_t* w1T     = (half_t*)bump((size_t)2 * DMODEL * MLP_H * 2);
    half_t* w2T     = (half_t*)bump((size_t)2 * MLP_H * DMODEL * 2);
    half_t* wheadT  = (half_t*)bump((size_t)DMODEL * HN_PAD * 2);
    float*  hd_out  = (float*)bump((size_t)N_TOK * HN_PAD * 4);
    (void)ws_size; (void)in_sizes; (void)n_in; (void)out_size;

    const int CT = 256, CG = 2048;
    convert_wT<<<CG, CT, 0, stream>>>(patch_w, wpT, DMODEL, KP, DMODEL, KP_PAD);
    for (int l = 0; l < 2; ++l) {
        convert_wT<<<CG, CT, 0, stream>>>(qkv_w + (size_t)l * TRI * DMODEL,
                                          wqkvT + (size_t)l * DMODEL * TRI, TRI, DMODEL, TRI, DMODEL);
        convert_wT<<<CG, CT, 0, stream>>>(proj_w + (size_t)l * DMODEL * DMODEL,
                                          wprojT + (size_t)l * DMODEL * DMODEL, DMODEL, DMODEL, DMODEL, DMODEL);
        convert_wT<<<CG, CT, 0, stream>>>(mlp_w1 + (size_t)l * MLP_H * DMODEL,
                                          w1T + (size_t)l * DMODEL * MLP_H, MLP_H, DMODEL, MLP_H, DMODEL);
        convert_wT<<<CG, CT, 0, stream>>>(mlp_w2 + (size_t)l * DMODEL * MLP_H,
                                          w2T + (size_t)l * MLP_H * DMODEL, DMODEL, MLP_H, DMODEL, MLP_H);
    }
    convert_wT<<<CG, CT, 0, stream>>>(head_w, wheadT, HN, DMODEL, HN_PAD, DMODEL);

    im2col_patch<<<CG, CT, 0, stream>>>(x, xp16);
    gemm_wmma<0><<<dim3(DMODEL / 128, N_TOK / 128), 256, 0, stream>>>(
        xp16, wpT, patch_b, DMODEL, h, nullptr, N_TOK, DMODEL, KP_PAD, KP_PAD, DMODEL, DMODEL);
    add_pos<<<CG, CT, 0, stream>>>(h, pos);

    for (int l = 0; l < 2; ++l) {
        ln_kernel<<<N_TOK, 128, 0, stream>>>(h, ln1_w + l * DMODEL, ln1_b + l * DMODEL, y16);
        gemm_wmma<1><<<dim3(TRI / 128, N_TOK / 128), 256, 0, stream>>>(
            y16, wqkvT + (size_t)l * DMODEL * TRI, qkv_b + l * TRI, TRI,
            nullptr, qkv16, N_TOK, TRI, DMODEL, DMODEL, TRI, TRI);
        flash_attn<<<dim3(N_TOK / 128, HEADS), 256, 0, stream>>>(qkv16, o16);
        gemm_wmma<2><<<dim3(DMODEL / 128, N_TOK / 128), 256, 0, stream>>>(
            o16, wprojT + (size_t)l * DMODEL * DMODEL, proj_b + l * DMODEL, DMODEL,
            h, nullptr, N_TOK, DMODEL, DMODEL, DMODEL, DMODEL, DMODEL);
        ln_kernel<<<N_TOK, 128, 0, stream>>>(h, ln2_w + l * DMODEL, ln2_b + l * DMODEL, y16);
        gemm_wmma<3><<<dim3(MLP_H / 128, N_TOK / 128), 256, 0, stream>>>(
            y16, w1T + (size_t)l * DMODEL * MLP_H, mlp_b1 + l * MLP_H, MLP_H,
            nullptr, g16, N_TOK, MLP_H, DMODEL, DMODEL, MLP_H, MLP_H);
        gemm_wmma<2><<<dim3(DMODEL / 128, N_TOK / 128), 256, 0, stream>>>(
            g16, w2T + (size_t)l * MLP_H * DMODEL, mlp_b2 + l * DMODEL, DMODEL,
            h, nullptr, N_TOK, DMODEL, MLP_H, MLP_H, DMODEL, DMODEL);
    }

    ln_kernel<<<N_TOK, 128, 0, stream>>>(h, norm_w, norm_b, y16);
    gemm_wmma<0><<<dim3(HN_PAD / 128, N_TOK / 128), 256, 0, stream>>>(
        y16, wheadT, head_b, HN, hd_out, nullptr, N_TOK, HN_PAD, DMODEL, DMODEL, HN_PAD, HN_PAD);
    head_reorder<<<CG, CT, 0, stream>>>(hd_out, out);
}
